// DGCNN_32040456028658
// MI455X (gfx1250) — compile-verified, matrix operand order checked by reading, not run on hardware
//
#include <hip/hip_runtime.h>
#include <hip/hip_bf16.h>
#include <float.h>

// ---------------------------------------------------------------------------
// DGCNN forward for MI455X (gfx1250, wave32, WMMA).
// All GEMM-like work mapped onto v_wmma_f32_16x16x32_bf16 (f32 accumulate).
// Entire intermediate working set (~41 MB) fits in the 192 MB L2 -> compute
// bound; WMMA bf16 is the throughput path.
// ---------------------------------------------------------------------------

typedef __attribute__((ext_vector_type(16))) __bf16       v16bf;
typedef __attribute__((ext_vector_type(8)))  float        v8f;
typedef __attribute__((ext_vector_type(8)))  unsigned int v8u;

#define NB    8
#define NPTS  2048
#define KNBR  40
#define CATW  512

__device__ __forceinline__ unsigned short f32_bf16(float f) {
  unsigned u = __builtin_bit_cast(unsigned, f);
  u += 0x7FFFu + ((u >> 16) & 1u);           // round-to-nearest-even
  return (unsigned short)(u >> 16);
}
__device__ __forceinline__ float lrelu(float v) { return v >= 0.f ? v : 0.2f * v; }

// --------------------------------------------------------------------------
// Weight convert: W[din][dout] f32 -> Wt[dout][K2] bf16, K zero-padded.
// Transposed so WMMA B-fragments load contiguous K pairs as 32-bit words.
// --------------------------------------------------------------------------
__global__ void wcvt_kernel(const float* __restrict__ W, unsigned short* __restrict__ Wt,
                            int din, int dout, int K2) {
  int e = blockIdx.x * blockDim.x + threadIdx.x;
  if (e >= dout * K2) return;
  int n = e / K2, k = e - n * K2;
  float v = (k < din) ? W[(size_t)k * dout + n] : 0.f;
  Wt[(size_t)n * K2 + k] = f32_bf16(v);
}

// --------------------------------------------------------------------------
// kNN: one 4-wave block handles 4 query rows of one batch.
// Exact f32 squared distances (incl. self = 0) staged in LDS, then 40 rounds
// of wave-wide arg-min selection with index tie-break.
// --------------------------------------------------------------------------
__global__ __launch_bounds__(128) void knn_kernel(const float* __restrict__ h, int hs, int C,
                                                  int* __restrict__ idxout) {
  __shared__ float sdist[4 * NPTS];      // 32 KB: 4 rows x 2048 distances
  __shared__ float colf[32 * 129];       // column-tile features, stride C+1
  __shared__ float rowf[4 * 129];        // query-row features
  const int tid  = threadIdx.x;
  const int wave = tid >> 5, lane = tid & 31;
  const int b    = blockIdx.x >> 9;      // NPTS/4 = 512 row groups per batch
  const int rg   = blockIdx.x & 511;
  const int row0 = rg * 4;
  const int Cp   = C + 1;                // odd stride -> conflict-free LDS

  for (int e = tid; e < 4 * C; e += 128) {
    int r = e / C, c = e - r * C;
    rowf[r * Cp + c] = h[(size_t)(b * NPTS + row0 + r) * hs + c];
  }
  for (int ct = 0; ct < NPTS / 32; ++ct) {
    __syncthreads();
    for (int e = tid; e < 32 * C; e += 128) {
      int cc = e / C, c = e - cc * C;
      colf[cc * Cp + c] = h[(size_t)(b * NPTS + ct * 32 + cc) * hs + c];
    }
    __syncthreads();
    float d = 0.f;
    const float* rp = rowf + wave * Cp;
    const float* cp = colf + lane * Cp;
    for (int c = 0; c < C; ++c) { float df = rp[c] - cp[c]; d += df * df; }
    sdist[wave * NPTS + ct * 32 + lane] = d;
  }
  __syncthreads();

  const int row = row0 + wave;
  float* dw = sdist + wave * NPTS;
  for (int kk = 0; kk < KNBR; ++kk) {
    float best = FLT_MAX; int bcol = lane;
    for (int t = 0; t < NPTS / 32; ++t) {
      float v = dw[t * 32 + lane];
      if (v < best) { best = v; bcol = t * 32 + lane; }
    }
    for (int off = 16; off > 0; off >>= 1) {       // wave32 arg-min reduce
      float ob = __shfl_xor(best, off, 32);
      int   oc = __shfl_xor(bcol, off, 32);
      if (ob < best || (ob == best && oc < bcol)) { best = ob; bcol = oc; }
    }
    if ((bcol & 31) == lane) dw[bcol] = FLT_MAX;   // winner lane masks element
    if (lane == 0) idxout[(size_t)(b * NPTS + row) * KNBR + kk] = bcol;
  }
}

// --------------------------------------------------------------------------
// EdgeConv: one 128-thread block per point.
// LDS edge matrix E (48 x K2, rows 40..47 zero-pad) in bf16, WMMA GEMM
// against Wt[dout][K2], bias + leaky-ReLU, max over the 40 valid rows.
// A-fragment K mapping (16-bit A 16x32): klo = kt*32 + 2q + (q>=4?8:0) + hi*8.
// --------------------------------------------------------------------------
__global__ __launch_bounds__(128) void edgeconv_kernel(
    const float* __restrict__ h, int hs, int C, int K2, int Cout,
    const int* __restrict__ idx, const unsigned short* __restrict__ Wt,
    const float* __restrict__ bias, float* __restrict__ out, int os) {
  __shared__ unsigned short Es[48 * 256];   // up to 24 KB bf16 edge features
  __shared__ float cenf[128];
  __shared__ int   nidx[KNBR];
  const int tid = threadIdx.x;
  const int p = blockIdx.x;
  const int b = p / NPTS, n = p - b * NPTS;
  const float* hc = h + (size_t)(b * NPTS + n) * hs;

  if (tid < KNBR) nidx[tid] = idx[(size_t)p * KNBR + tid];
  for (int c = tid; c < C; c += 128) cenf[c] = hc[c];
  __syncthreads();

  const int C2 = 2 * C;
  for (int e = tid; e < 48 * K2; e += 128) {
    int kk = e / K2, k = e - kk * K2;
    float v = 0.f;
    if (kk < KNBR) {
      if (k < C)        v = h[(size_t)(b * NPTS + nidx[kk]) * hs + k] - cenf[k];
      else if (k < C2)  v = cenf[k - C];
    }
    Es[kk * K2 + k] = f32_bf16(v);
  }
  __syncthreads();

  const int wave = tid >> 5, lane = tid & 31;
  const int lm = lane & 15, hi = lane >> 4;
  const int ntiles = Cout >> 4, ktiles = K2 >> 5;

  for (int nt = wave; nt < ntiles; nt += 4) {
    v8f acc0 = {}, acc1 = {}, acc2 = {};
    const unsigned short* wp = Wt + (size_t)(nt * 16 + lm) * K2;
    for (int kt = 0; kt < ktiles; ++kt) {
      v8u bu, a0, a1, a2;
      const int kbase = kt * 32 + hi * 8;
#pragma unroll
      for (int q = 0; q < 8; ++q) {
        const int klo = kbase + 2 * q + ((q >= 4) ? 8 : 0);
        bu[q] = *(const unsigned int*)(wp + klo);
        a0[q] = *(const unsigned int*)(Es + (lm)      * K2 + klo);
        a1[q] = *(const unsigned int*)(Es + (16 + lm) * K2 + klo);
        a2[q] = *(const unsigned int*)(Es + (32 + lm) * K2 + klo);
      }
      const v16bf bb = __builtin_bit_cast(v16bf, bu);
      acc0 = __builtin_amdgcn_wmma_f32_16x16x32_bf16(false, __builtin_bit_cast(v16bf, a0),
                                                     false, bb, (short)0, acc0, false, false);
      acc1 = __builtin_amdgcn_wmma_f32_16x16x32_bf16(false, __builtin_bit_cast(v16bf, a1),
                                                     false, bb, (short)0, acc1, false, false);
      acc2 = __builtin_amdgcn_wmma_f32_16x16x32_bf16(false, __builtin_bit_cast(v16bf, a2),
                                                     false, bb, (short)0, acc2, false, false);
    }
    // epilogue: bias + leaky-ReLU, max over k (rows 40..47 are padding).
    const float bv = bias[nt * 16 + lm];
    float m = -FLT_MAX;
#pragma unroll
    for (int r = 0; r < 8; ++r) {
      m = fmaxf(m, lrelu(acc0[r] + bv));           // rows 0..15
      m = fmaxf(m, lrelu(acc1[r] + bv));           // rows 16..31
      if (!hi) m = fmaxf(m, lrelu(acc2[r] + bv));  // rows 32..39 (lanes<16 only)
    }
    m = fmaxf(m, __shfl_xor(m, 16, 32));           // combine row-halves
    if (!hi) out[(size_t)(b * NPTS + n) * os + nt * 16 + lm] = m;
  }
}

// --------------------------------------------------------------------------
// Final: cat(B,N,512) @ Wf(512,1024) + bf, per-tile max over rows -> partials.
// One WMMA 16x16 tile per wave, K-loop of 16; A converted f32->bf16 on load.
// --------------------------------------------------------------------------
__global__ __launch_bounds__(128) void fingemm_kernel(
    const float* __restrict__ cat, const unsigned short* __restrict__ Wft,
    const float* __restrict__ bf, float* __restrict__ part) {
  const int wave = threadIdx.x >> 5, lane = threadIdx.x & 31;
  const int lm = lane & 15, hi = lane >> 4;
  const int t  = blockIdx.x * 4 + wave;      // 8*128*64 tiles total
  const int b  = t >> 13;
  const int mt = (t >> 6) & 127;
  const int nt = t & 63;

  v8f acc = {};
  const unsigned short* wp = Wft + (size_t)(nt * 16 + lm) * CATW;
  const float* ap = cat + ((size_t)b * NPTS + mt * 16 + lm) * CATW;
  for (int kt = 0; kt < CATW / 32; ++kt) {
    v8u au, bu;
    const int kbase = kt * 32 + hi * 8;
#pragma unroll
    for (int q = 0; q < 8; ++q) {
      const int klo = kbase + 2 * q + ((q >= 4) ? 8 : 0);
      bu[q] = *(const unsigned int*)(wp + klo);
      float2 f2 = *(const float2*)(ap + klo);
      au[q] = (unsigned)f32_bf16(f2.x) | ((unsigned)f32_bf16(f2.y) << 16);
    }
    acc = __builtin_amdgcn_wmma_f32_16x16x32_bf16(false, __builtin_bit_cast(v16bf, au),
                                                  false, __builtin_bit_cast(v16bf, bu),
                                                  (short)0, acc, false, false);
  }
  const float bv = bf[nt * 16 + lm];
  float m = -FLT_MAX;
#pragma unroll
  for (int r = 0; r < 8; ++r) m = fmaxf(m, acc[r] + bv);
  m = fmaxf(m, __shfl_xor(m, 16, 32));
  if (!hi) part[((size_t)b * 128 + mt) * 1024 + nt * 16 + lm] = m;
}

__global__ void finreduce_kernel(const float* __restrict__ part, float* __restrict__ out) {
  int e = blockIdx.x * blockDim.x + threadIdx.x;
  if (e >= NB * 1024) return;
  int b = e >> 10, n = e & 1023;
  const float* pp = part + (size_t)b * 128 * 1024 + n;
  float m = -FLT_MAX;
  for (int i = 0; i < 128; ++i) m = fmaxf(m, pp[(size_t)i * 1024]);
  out[e] = m;
}

// --------------------------------------------------------------------------
extern "C" void kernel_launch(void* const* d_in, const int* in_sizes, int n_in,
                              void* d_out, int out_size, void* d_ws, size_t ws_size,
                              hipStream_t stream) {
  const float* x   = (const float*)d_in[0];
  const float* W0  = (const float*)d_in[1];  const float* b0 = (const float*)d_in[2];
  const float* W1  = (const float*)d_in[3];  const float* b1 = (const float*)d_in[4];
  const float* W2  = (const float*)d_in[5];  const float* b2 = (const float*)d_in[6];
  const float* W3  = (const float*)d_in[7];  const float* b3 = (const float*)d_in[8];
  const float* Wf  = (const float*)d_in[9];  const float* bfv = (const float*)d_in[10];

  char* w = (char*)d_ws;
  auto carve = [&](size_t bytes) -> void* {
    void* p = (void*)w; w += (bytes + 255) & ~(size_t)255; return p;
  };
  float* cat = (float*)carve((size_t)NB * NPTS * CATW * 4);   // 32 MB feature concat
  int*   idx = (int*)carve((size_t)NB * NPTS * KNBR * 4);     // 2.6 MB (reused per layer)
  unsigned short* Wt0 = (unsigned short*)carve((size_t)64   * 32  * 2);
  unsigned short* Wt1 = (unsigned short*)carve((size_t)64   * 128 * 2);
  unsigned short* Wt2 = (unsigned short*)carve((size_t)128  * 128 * 2);
  unsigned short* Wt3 = (unsigned short*)carve((size_t)256  * 256 * 2);
  unsigned short* Wtf = (unsigned short*)carve((size_t)1024 * CATW * 2);
  float* part = (float*)carve((size_t)NB * 128 * 1024 * 4);   // 4 MB partial maxima

  // Weight transpose/convert to bf16 (K padded to multiple of 32).
  wcvt_kernel<<<(64   * 32  + 255) / 256, 256, 0, stream>>>(W0, Wt0, 6,   64,   32);
  wcvt_kernel<<<(64   * 128 + 255) / 256, 256, 0, stream>>>(W1, Wt1, 128, 64,   128);
  wcvt_kernel<<<(128  * 128 + 255) / 256, 256, 0, stream>>>(W2, Wt2, 128, 128,  128);
  wcvt_kernel<<<(256  * 256 + 255) / 256, 256, 0, stream>>>(W3, Wt3, 256, 256,  256);
  wcvt_kernel<<<(1024 * 512 + 255) / 256, 256, 0, stream>>>(Wf, Wtf, 512, 1024, 512);

  const int knn_grid = NB * (NPTS / 4);
  const int ec_grid  = NB * NPTS;

  // Layer 0: h = x (C=3), K2 padded 6->32, Cout=64 -> cat[...,0:64]
  knn_kernel<<<knn_grid, 128, 0, stream>>>(x, 3, 3, idx);
  edgeconv_kernel<<<ec_grid, 128, 0, stream>>>(x, 3, 3, 32, 64, idx, Wt0, b0, cat + 0, CATW);
  // Layer 1: C=64, K2=128, Cout=64 -> cat[...,64:128]
  knn_kernel<<<knn_grid, 128, 0, stream>>>(cat + 0, CATW, 64, idx);
  edgeconv_kernel<<<ec_grid, 128, 0, stream>>>(cat + 0, CATW, 64, 128, 64, idx, Wt1, b1, cat + 64, CATW);
  // Layer 2: C=64, K2=128, Cout=128 -> cat[...,128:256]
  knn_kernel<<<knn_grid, 128, 0, stream>>>(cat + 64, CATW, 64, idx);
  edgeconv_kernel<<<ec_grid, 128, 0, stream>>>(cat + 64, CATW, 64, 128, 128, idx, Wt2, b2, cat + 128, CATW);
  // Layer 3: C=128, K2=256, Cout=256 -> cat[...,256:512]
  knn_kernel<<<knn_grid, 128, 0, stream>>>(cat + 128, CATW, 128, idx);
  edgeconv_kernel<<<ec_grid, 128, 0, stream>>>(cat + 128, CATW, 128, 256, 256, idx, Wt3, b3, cat + 256, CATW);

  // Final linear + max over N.
  fingemm_kernel<<<(NB * 128 * 64) / 4, 128, 0, stream>>>(cat, Wtf, bfv, part);
  finreduce_kernel<<<(NB * 1024 + 255) / 256, 256, 0, stream>>>(part, (float*)d_out);
}
